// TVConvInvertedResidual_34909494182031
// MI455X (gfx1250) — compile-verified, matrix operand order checked by reading, not run on hardware
//
#include <hip/hip_runtime.h>

// ---------------- problem constants (from reference) ----------------
#define HWD   56
#define NPIX  3136          // 56*56
#define BSZ   16
#define INPC  64
#define OUPC  64
#define HID   384
#define ICC   64
#define PCH   4
#define K2    9
#define KF    576           // ICC * 9
#define NF    3456          // K2 * HID
#define EPSV  1e-5f

typedef __attribute__((ext_vector_type(16))) __bf16 v16bf;
typedef __attribute__((ext_vector_type(8)))  __bf16 v8bf;
typedef __attribute__((ext_vector_type(8)))  float  v8f;

// Build one 16x16x32 bf16 WMMA operand fragment for this lane.
// Row-major [row][K]; `p` points at row + kstep*32 + koff, koff = (lane<16?0:8).
// Lane elements e=0..7 hold K=koff+e, e=8..15 hold K=koff+16+(e-8):
// two contiguous 16B groups separated by 16 bf16.
__device__ __forceinline__ v16bf load_frag(const __bf16* p) {
  v8bf lo = *(const v8bf*)(p);
  v8bf hi = *(const v8bf*)(p + 16);
  v16bf r;
#pragma unroll
  for (int i = 0; i < 8; ++i) { r[i] = lo[i]; r[i + 8] = hi[i]; }
  return r;
}

// ---------------- weight fp32 -> bf16 conversion ----------------
__global__ void cvt_weights_kernel(const float* __restrict__ wexp,
                                   const float* __restrict__ wproj,
                                   const float* __restrict__ wconvf,
                                   __bf16* __restrict__ wexp_bf,
                                   __bf16* __restrict__ wproj_bf,
                                   __bf16* __restrict__ wconvf_bf) {
  int i = blockIdx.x * blockDim.x + threadIdx.x;
  if (i < NF * KF)     wconvf_bf[i] = (__bf16)wconvf[i];
  if (i < HID * INPC)  wexp_bf[i]   = (__bf16)wexp[i];
  if (i < OUPC * HID)  wproj_bf[i]  = (__bf16)wproj[i];
}

// ---------------- expand 1x1 conv + BN1 + ReLU6 (WMMA) ----------------
// One workgroup per pixel. M = batch(16), N = HID(384), K = INP(64).
// Output h is NHWC: hbuf[(b*NPIX + p)*HID + c]  -> lane-contiguous stores.
__global__ __launch_bounds__(128) void expand_kernel(
    const float* __restrict__ x, const __bf16* __restrict__ wexp_bf,
    const float* __restrict__ g1, const float* __restrict__ b1,
    const float* __restrict__ m1, const float* __restrict__ v1,
    float* __restrict__ hbuf) {
  const int p    = blockIdx.x;
  const int tid  = threadIdx.x;
  const int lane = tid & 31;
  const int wave = tid >> 5;
  __shared__ __align__(16) __bf16 Abuf[BSZ * INPC];     // [M=16][K=64]

  for (int idx = tid; idx < BSZ * INPC; idx += 128)
    Abuf[idx] = (__bf16)x[idx * NPIX + p];              // idx = b*INPC + i
  __syncthreads();

  const int mrow = lane & 15;
  const int koff = (lane < 16) ? 0 : 8;
  const v16bf a0 = load_frag(&Abuf[mrow * INPC + 0  + koff]);
  const v16bf a1 = load_frag(&Abuf[mrow * INPC + 32 + koff]);

  for (int t = wave; t < HID / 16; t += 4) {            // 24 N-tiles / 4 waves
    const int n = t * 16 + mrow;
    v8f c = {};
    v16bf bb0 = load_frag(&wexp_bf[n * INPC + 0  + koff]);
    v16bf bb1 = load_frag(&wexp_bf[n * INPC + 32 + koff]);
    c = __builtin_amdgcn_wmma_f32_16x16x32_bf16(false, a0, false, bb0, (short)0, c, false, false);
    c = __builtin_amdgcn_wmma_f32_16x16x32_bf16(false, a1, false, bb1, (short)0, c, false, false);

    const float sc = g1[n] * rsqrtf(v1[n] + EPSV);
    const float sh = b1[n] - m1[n] * sc;
#pragma unroll
    for (int r = 0; r < 8; ++r) {
      const int bb = r + ((lane < 16) ? 0 : 8);         // batch index (M row)
      float o = c[r] * sc + sh;
      o = fminf(fmaxf(o, 0.f), 6.f);
      hbuf[((size_t)bb * NPIX + p) * HID + n] = o;      // NHWC, coalesced in n
    }
  }
}

// ---------------- small 3x3 conv + LN partial sums ----------------
__global__ __launch_bounds__(256) void conv3x3_kernel(
    const float* __restrict__ in, const float* __restrict__ w, int cin,
    float* __restrict__ outpre, float* __restrict__ partials) {
  const int co = blockIdx.y;
  const int p  = blockIdx.x * 256 + threadIdx.x;
  __shared__ float wsm[ICC * K2];
  __shared__ float red1[256];
  __shared__ float red2[256];
  for (int i = threadIdx.x; i < cin * K2; i += 256)
    wsm[i] = w[co * cin * K2 + i];
  __syncthreads();

  float acc = 0.f;
  if (p < NPIX) {
    const int hh = p / HWD, ww = p % HWD;
    for (int ci = 0; ci < cin; ++ci) {
      const float* inc = in + ci * NPIX;
#pragma unroll
      for (int t = 0; t < 9; ++t) {
        const int hy = hh + t / 3 - 1, wx = ww + t % 3 - 1;
        if (hy >= 0 && hy < HWD && wx >= 0 && wx < HWD)
          acc += wsm[ci * K2 + t] * inc[hy * HWD + wx];
      }
    }
    outpre[co * NPIX + p] = acc;
  }
  const float a1 = (p < NPIX) ? acc : 0.f;
  red1[threadIdx.x] = a1;
  red2[threadIdx.x] = a1 * a1;
  __syncthreads();
  for (int s = 128; s > 0; s >>= 1) {
    if (threadIdx.x < s) {
      red1[threadIdx.x] += red1[threadIdx.x + s];
      red2[threadIdx.x] += red2[threadIdx.x + s];
    }
    __syncthreads();
  }
  if (threadIdx.x == 0) {
    const int bid = co * gridDim.x + blockIdx.x;
    partials[2 * bid]     = red1[0];
    partials[2 * bid + 1] = red2[0];
  }
}

// ---------------- deterministic LN stats reduction ----------------
__global__ __launch_bounds__(256) void ln_stats_kernel(
    const float* __restrict__ partials, int nblk, float* __restrict__ stats) {
  __shared__ float r1[256], r2[256];
  float s1 = 0.f, s2 = 0.f;
  for (int i = threadIdx.x; i < nblk; i += 256) { s1 += partials[2 * i]; s2 += partials[2 * i + 1]; }
  r1[threadIdx.x] = s1; r2[threadIdx.x] = s2;
  __syncthreads();
  for (int s = 128; s > 0; s >>= 1) {
    if (threadIdx.x < s) { r1[threadIdx.x] += r1[threadIdx.x + s]; r2[threadIdx.x] += r2[threadIdx.x + s]; }
    __syncthreads();
  }
  if (threadIdx.x == 0) {
    const float N = (float)(ICC * NPIX);
    const float mean = r1[0] / N;
    const float var  = r2[0] / N - mean * mean;
    stats[0] = mean;
    stats[1] = rsqrtf(var + EPSV);
  }
}

// ---------------- LN affine + ReLU ----------------
__global__ void ln_apply_kernel(const float* __restrict__ pre, const float* __restrict__ g,
                                const float* __restrict__ b, const float* __restrict__ stats,
                                float* __restrict__ out) {
  const int i = blockIdx.x * blockDim.x + threadIdx.x;
  if (i < ICC * NPIX) {
    const float yn = (pre[i] - stats[0]) * stats[1] * g[i] + b[i];
    out[i] = fmaxf(yn, 0.f);
  }
}

// ---------------- final weight-gen conv as implicit GEMM (WMMA) ----------------
// One workgroup per 16-pixel M-tile. M = pixels(16), N = 3456, K = 576 (c*9+tap).
// Output wgt is [p][n] -> lane-contiguous stores, contiguous tap reads later.
__global__ __launch_bounds__(128) void convf_kernel(
    const float* __restrict__ y, const __bf16* __restrict__ wf_bf,
    float* __restrict__ wgt) {
  const int p0   = blockIdx.x * 16;
  const int tid  = threadIdx.x;
  const int lane = tid & 31;
  const int wave = tid >> 5;
  __shared__ __align__(16) __bf16 Abuf[16 * KF];        // 18 KB, [M=16][K=576]

  for (int idx = tid; idx < 16 * KF; idx += 128) {      // im2col staging
    const int mm = idx / KF;
    const int kk = idx - mm * KF;
    const int ci = kk / K2, t = kk - ci * K2;
    const int p  = p0 + mm;
    const int hy = p / HWD + t / 3 - 1;
    const int wx = p % HWD + t % 3 - 1;
    float v = 0.f;
    if (hy >= 0 && hy < HWD && wx >= 0 && wx < HWD) v = y[ci * NPIX + hy * HWD + wx];
    Abuf[mm * KF + kk] = (__bf16)v;
  }
  __syncthreads();

  const int mrow = lane & 15;
  const int koff = (lane < 16) ? 0 : 8;
  for (int nt = wave; nt < NF / 16; nt += 4) {          // 216 N-tiles / 4 waves
    const int n = nt * 16 + mrow;
    const __bf16* brow = wf_bf + (size_t)n * KF;
    if (nt + 4 < NF / 16) __builtin_prefetch(brow + (size_t)64 * KF, 0, 1);
    v8f c = {};
#pragma unroll 2
    for (int s = 0; s < KF / 32; ++s) {                 // 18 K-steps, 2x unrolled
      v16bf a  = load_frag(&Abuf[mrow * KF + s * 32 + koff]);
      v16bf bv = load_frag(&brow[s * 32 + koff]);
      c = __builtin_amdgcn_wmma_f32_16x16x32_bf16(false, a, false, bv, (short)0, c, false, false);
    }
#pragma unroll
    for (int r = 0; r < 8; ++r) {
      const int mm = r + ((lane < 16) ? 0 : 8);
      wgt[(size_t)(p0 + mm) * NF + n] = c[r];           // [p][n], coalesced in n
    }
  }
}

// ---------------- fused unfold*wgt + BN2 + ReLU6 + project + BN3 + residual ----------------
// One workgroup per pixel. Unfold-sum builds A (M=batch16, K=HID) in LDS,
// then WMMA against w_proj (N=64), epilogue adds BN3 + residual.
__global__ __launch_bounds__(128) void tvproj_kernel(
    const float* __restrict__ hbuf, const float* __restrict__ wgt,
    const float* __restrict__ x, const __bf16* __restrict__ wproj_bf,
    const float* __restrict__ g2, const float* __restrict__ b2,
    const float* __restrict__ m2, const float* __restrict__ v2,
    const float* __restrict__ g3, const float* __restrict__ b3,
    const float* __restrict__ m3, const float* __restrict__ v3,
    float* __restrict__ out) {
  const int p    = blockIdx.x;
  const int hh   = p / HWD, ww = p % HWD;
  const int tid  = threadIdx.x;
  const int lane = tid & 31;
  const int wave = tid >> 5;
  __shared__ __align__(16) __bf16 Abuf[BSZ * HID];      // 12 KB, [M=16][K=384]

  for (int c = tid; c < HID; c += 128) {                // threads = consecutive channels
    float wk[9];
#pragma unroll
    for (int t = 0; t < 9; ++t) wk[t] = wgt[(size_t)p * NF + c * K2 + t];   // contiguous
    const float sc = g2[c] * rsqrtf(v2[c] + EPSV);
    const float sh = b2[c] - m2[c] * sc;
    for (int bb = 0; bb < BSZ; ++bb) {
      const float* hb = hbuf + (size_t)bb * NPIX * HID + c;   // NHWC: +pn*HID
      float acc = 0.f;
#pragma unroll
      for (int t = 0; t < 9; ++t) {
        const int hy = hh + t / 3 - 1, wx = ww + t % 3 - 1;
        if (hy >= 0 && hy < HWD && wx >= 0 && wx < HWD)
          acc += wk[t] * hb[(size_t)(hy * HWD + wx) * HID];   // coalesced across threads
      }
      const float tv = fminf(fmaxf(acc * sc + sh, 0.f), 6.f);
      Abuf[bb * HID + c] = (__bf16)tv;
    }
  }
  __syncthreads();

  const int mrow = lane & 15;
  const int koff = (lane < 16) ? 0 : 8;
  const int n    = wave * 16 + mrow;                    // 4 waves = 4 N-tiles = OUP 64
  const __bf16* brow = wproj_bf + n * HID;
  v8f c = {};
#pragma unroll 2
  for (int s = 0; s < HID / 32; ++s) {                  // 12 K-steps, 2x unrolled
    v16bf a  = load_frag(&Abuf[mrow * HID + s * 32 + koff]);
    v16bf bv = load_frag(&brow[s * 32 + koff]);
    c = __builtin_amdgcn_wmma_f32_16x16x32_bf16(false, a, false, bv, (short)0, c, false, false);
  }
  const float sc3 = g3[n] * rsqrtf(v3[n] + EPSV);
  const float sh3 = b3[n] - m3[n] * sc3;
#pragma unroll
  for (int r = 0; r < 8; ++r) {
    const int bb = r + ((lane < 16) ? 0 : 8);
    const size_t oi = ((size_t)(bb * OUPC + n)) * NPIX + p;   // NCHW output (fixed)
    out[oi] = x[oi] + (c[r] * sc3 + sh3);
  }
}

// ---------------- host launcher ----------------
extern "C" void kernel_launch(void* const* d_in, const int* in_sizes, int n_in,
                              void* d_out, int out_size, void* d_ws, size_t ws_size,
                              hipStream_t stream) {
  (void)in_sizes; (void)n_in; (void)out_size; (void)ws_size;
  const float* x        = (const float*)d_in[0];
  const float* w_exp    = (const float*)d_in[1];
  const float* bn1_g    = (const float*)d_in[2];
  const float* bn1_b    = (const float*)d_in[3];
  const float* bn1_m    = (const float*)d_in[4];
  const float* bn1_v    = (const float*)d_in[5];
  const float* posi_map = (const float*)d_in[6];
  const float* wl_conv0 = (const float*)d_in[7];
  const float* wl_ln0_g = (const float*)d_in[8];
  const float* wl_ln0_b = (const float*)d_in[9];
  const float* wl_convm = (const float*)d_in[10];
  const float* wl_lnm_g = (const float*)d_in[11];
  const float* wl_lnm_b = (const float*)d_in[12];
  const float* wl_convf = (const float*)d_in[13];
  const float* bn2_g    = (const float*)d_in[14];
  const float* bn2_b    = (const float*)d_in[15];
  const float* bn2_m    = (const float*)d_in[16];
  const float* bn2_v    = (const float*)d_in[17];
  const float* w_proj   = (const float*)d_in[18];
  const float* bn3_g    = (const float*)d_in[19];
  const float* bn3_b    = (const float*)d_in[20];
  const float* bn3_m    = (const float*)d_in[21];
  const float* bn3_v    = (const float*)d_in[22];
  float* out = (float*)d_out;

  // workspace carve-up (256B aligned); total ~126 MB (L2-resident on MI455X)
  char* ws = (char*)d_ws;
  size_t off = 0;
  auto carve = [&](size_t bytes) -> char* {
    char* p = ws + off;
    off = (off + bytes + 255) & ~(size_t)255;
    return p;
  };
  float*  hbuf     = (float*)carve((size_t)BSZ * HID * NPIX * 4);   // 77.1 MB, NHWC
  float*  wgtbuf   = (float*)carve((size_t)NF * NPIX * 4);          // 43.4 MB, [p][n]
  float*  ybuf0    = (float*)carve((size_t)ICC * NPIX * 4);         // pre-LN
  float*  ybuf1    = (float*)carve((size_t)ICC * NPIX * 4);         // post-LN
  __bf16* wexp_bf  = (__bf16*)carve((size_t)HID * INPC * 2);
  __bf16* wproj_bf = (__bf16*)carve((size_t)OUPC * HID * 2);
  __bf16* wf_bf    = (__bf16*)carve((size_t)NF * KF * 2);           // 4.0 MB
  float*  partials = (float*)carve((size_t)2 * 64 * 16 * 4);
  float*  stats    = (float*)carve(256);

  const int PXB  = (NPIX + 255) / 256;    // 13 pixel-blocks for conv3x3
  const int NBLK = 64 * PXB;              // LN partial count

  // 0) weight conversion fp32 -> bf16
  {
    const int n = NF * KF;
    cvt_weights_kernel<<<(n + 255) / 256, 256, 0, stream>>>(
        w_exp, w_proj, wl_convf, wexp_bf, wproj_bf, wf_bf);
  }

  // 1) expand + BN1 + ReLU6  (WMMA, one workgroup per pixel)
  expand_kernel<<<NPIX, 128, 0, stream>>>(x, wexp_bf, bn1_g, bn1_b, bn1_m, bn1_v, hbuf);

  // 2) weight-generation net: conv0 + LN + ReLU, then 3x (convm + LN + ReLU)
  conv3x3_kernel<<<dim3(PXB, 64), 256, 0, stream>>>(posi_map, wl_conv0, PCH, ybuf0, partials);
  ln_stats_kernel<<<1, 256, 0, stream>>>(partials, NBLK, stats);
  ln_apply_kernel<<<(ICC * NPIX + 255) / 256, 256, 0, stream>>>(ybuf0, wl_ln0_g, wl_ln0_b, stats, ybuf1);
  for (int i = 0; i < 3; ++i) {
    conv3x3_kernel<<<dim3(PXB, 64), 256, 0, stream>>>(
        ybuf1, wl_convm + (size_t)i * ICC * ICC * K2, ICC, ybuf0, partials);
    ln_stats_kernel<<<1, 256, 0, stream>>>(partials, NBLK, stats);
    ln_apply_kernel<<<(ICC * NPIX + 255) / 256, 256, 0, stream>>>(
        ybuf0, wl_lnm_g + (size_t)i * ICC * NPIX, wl_lnm_b + (size_t)i * ICC * NPIX, stats, ybuf1);
  }

  // 3) final weight-gen conv (implicit GEMM, WMMA): y -> wgt [p][n]
  convf_kernel<<<NPIX / 16, 128, 0, stream>>>(ybuf1, wf_bf, wgtbuf);

  // 4) fused unfold*wgt + BN2 + ReLU6 + project (WMMA) + BN3 + residual
  tvproj_kernel<<<NPIX, 128, 0, stream>>>(hbuf, wgtbuf, x, wproj_bf,
                                          bn2_g, bn2_b, bn2_m, bn2_v,
                                          bn3_g, bn3_b, bn3_m, bn3_v, out);
}
